// GNModule_36189394436505
// MI455X (gfx1250) — matrix operand order repaired and checked by reading, not compiled, not run on hardware
//
#include <hip/hip_runtime.h>
#include <hip/hip_bf16.h>
#include <math.h>

#define NODES 100000
#define DIM   128
#define BN_EPS 1e-5f

typedef __attribute__((ext_vector_type(16))) __bf16 v16bf;
typedef __attribute__((ext_vector_type(8)))  float  v8f;

// ---------------- zero scratch ----------------
__global__ void zero_f4(float4* __restrict__ p, long long n4) {
    long long i = (long long)blockIdx.x * blockDim.x + threadIdx.x;
    long long stride = (long long)gridDim.x * blockDim.x;
    float4 z; z.x = z.y = z.z = z.w = 0.f;
    for (; i < n4; i += stride) p[i] = z;
}

// ---------------- degree (count by src) ----------------
__global__ void deg_kernel(const int* __restrict__ src, float* __restrict__ deg, int E) {
    int i = blockIdx.x * blockDim.x + threadIdx.x;
    int stride = gridDim.x * blockDim.x;
    for (; i < E; i += stride) atomicAdd(&deg[src[i]], 1.0f);
}

__global__ void dinv_kernel(const float* __restrict__ deg, float* __restrict__ dinv, int n) {
    int i = blockIdx.x * blockDim.x + threadIdx.x;
    if (i < n) { float d = deg[i]; dinv[i] = (d > 0.f) ? rsqrtf(d) : 0.f; }
}

// ---------------- sparse matvec: vout[dst] += norm * vin[src], one wave per edge ----------------
__global__ __launch_bounds__(256) void spmv_kernel(
    const int* __restrict__ src, const int* __restrict__ dst,
    const float* __restrict__ dinv,
    const float* __restrict__ vin, float* __restrict__ vout, int E) {
    int lane = threadIdx.x & 31;
    long long w  = ((long long)blockIdx.x * blockDim.x + threadIdx.x) >> 5;
    long long nw = ((long long)gridDim.x * blockDim.x) >> 5;
    for (long long e = w; e < E; e += nw) {
        int s = src[e], d = dst[e];
        float nrm = -(dinv[s] * dinv[d]);
        float4 v = ((const float4*)(vin + (long long)s * DIM))[lane];
        float* o = vout + (long long)d * DIM + lane * 4;
        atomicAdd(o + 0, nrm * v.x);
        atomicAdd(o + 1, nrm * v.y);
        atomicAdd(o + 2, nrm * v.z);
        atomicAdd(o + 3, nrm * v.w);
    }
}

// ---------------- W[hop][k][n] fp32 -> Wt[hop][n][k] bf16 ----------------
__global__ void prep_w(const float* __restrict__ W, __bf16* __restrict__ Wt) {
    int i = blockIdx.x * blockDim.x + threadIdx.x;
    if (i < 3 * DIM * DIM) {
        int hop = i / (DIM * DIM);
        int r   = i % (DIM * DIM);
        int n = r / DIM, k = r % DIM;
        Wt[i] = (__bf16)W[hop * DIM * DIM + k * DIM + n];
    }
}

// ---------------- fused Cheb GEMM + bias + PReLU + BN-stat accumulation ----------------
// blockDim = 128 (4 waves). Wave w computes 16-row strip (blockIdx.x*4 + w) x all 128 cols.
// out may alias tx1: each wave reads only its own rows before writing them.
__global__ __launch_bounds__(128) void cheb_gemm(
    const float* __restrict__ x, const float* __restrict__ tx1,
    const float* __restrict__ h2, const __bf16* __restrict__ Wt,
    const float* __restrict__ bias, const float* __restrict__ a_prelu,
    float* __restrict__ out, float* __restrict__ gsum, float* __restrict__ gsq) {
    __shared__ float s_sum[DIM];
    __shared__ float s_sq[DIM];
    int t = threadIdx.x;
    s_sum[t] = 0.f; s_sq[t] = 0.f;
    __syncthreads();

    int wave = t >> 5, lane = t & 31;
    int half = lane >> 4, lr = lane & 15;
    int strip = blockIdx.x * 4 + wave;
    bool active = (strip * 16 < NODES);      // wave-uniform

    if (active) {
        float ap = a_prelu[0];
        long long row = (long long)(strip * 16 + lr);
        const float* xr  = x   + row * DIM;
        const float* t1r = tx1 + row * DIM;
        const float* h2r = h2  + row * DIM;

        v8f acc[8] = {};
        #pragma unroll
        for (int hop = 0; hop < 3; hop++) {
            #pragma unroll
            for (int kt = 0; kt < 4; kt++) {
                // A fragment (16x32 bf16): lane holds row (lr); per ISA layout the 16
                // elements are two contiguous 8-float runs at kt*32+half*8 and +16.
                union { v16bf v; __bf16 b[16]; } af;
                int b0 = kt * 32 + half * 8;
                #pragma unroll
                for (int g = 0; g < 2; g++) {
                    int kk = b0 + g * 16;
                    float4 f0, f1;
                    if (hop == 0) {
                        f0 = *(const float4*)(xr + kk);
                        f1 = *(const float4*)(xr + kk + 4);
                    } else if (hop == 1) {
                        f0 = *(const float4*)(t1r + kk);
                        f1 = *(const float4*)(t1r + kk + 4);
                    } else {  // tx2 = 2*h2 - x on the fly
                        float4 h0 = *(const float4*)(h2r + kk);
                        float4 h1 = *(const float4*)(h2r + kk + 4);
                        float4 x0 = *(const float4*)(xr + kk);
                        float4 x1 = *(const float4*)(xr + kk + 4);
                        f0.x = 2.f * h0.x - x0.x; f0.y = 2.f * h0.y - x0.y;
                        f0.z = 2.f * h0.z - x0.z; f0.w = 2.f * h0.w - x0.w;
                        f1.x = 2.f * h1.x - x1.x; f1.y = 2.f * h1.y - x1.y;
                        f1.z = 2.f * h1.z - x1.z; f1.w = 2.f * h1.w - x1.w;
                    }
                    // native fptrunc -> bf16 (RNE) => packed v_cvt_pk_bf16_f32
                    af.b[g * 8 + 0] = (__bf16)f0.x; af.b[g * 8 + 1] = (__bf16)f0.y;
                    af.b[g * 8 + 2] = (__bf16)f0.z; af.b[g * 8 + 3] = (__bf16)f0.w;
                    af.b[g * 8 + 4] = (__bf16)f1.x; af.b[g * 8 + 5] = (__bf16)f1.y;
                    af.b[g * 8 + 6] = (__bf16)f1.z; af.b[g * 8 + 7] = (__bf16)f1.w;
                }
                #pragma unroll
                for (int nt = 0; nt < 8; nt++) {
                    // B fragment (32x16 bf16): lane holds col nt*16+lr, K = kt*32 + half*8 + e (e<8), kt*32 + 16 + half*8 + (e-8) (e>=8)
                    union { v16bf v; uint4 q[2]; } bf;
                    const uint4* bp = (const uint4*)(
                        Wt + ((hop * DIM + nt * 16 + lr) * DIM + kt * 32 + half * 8));
                    bf.q[0] = bp[0]; bf.q[1] = bp[2];
                    acc[nt] = __builtin_amdgcn_wmma_f32_16x16x32_bf16(
                        false, af.v, false, bf.v, (short)0, acc[nt], false, false);
                }
            }
        }
        // epilogue: bias + PReLU + store + per-column partial stats
        #pragma unroll
        for (int nt = 0; nt < 8; nt++) {
            int col = nt * 16 + lr;
            float b = bias[col];
            float ls = 0.f, lq = 0.f;
            #pragma unroll
            for (int r = 0; r < 8; r++) {
                float v = acc[nt][r] + b;
                v = (v >= 0.f) ? v : ap * v;
                out[(long long)(strip * 16 + half * 8 + r) * DIM + col] = v;
                ls += v; lq += v * v;
            }
            atomicAdd(&s_sum[col], ls);
            atomicAdd(&s_sq[col],  lq);
        }
    }
    __syncthreads();
    atomicAdd(&gsum[t], s_sum[t]);
    atomicAdd(&gsq[t],  s_sq[t]);
}

// ---------------- BN finalize (1 block, 128 threads) ----------------
__global__ void bn_finalize(const float* __restrict__ gsum, const float* __restrict__ gsq,
                            const float* __restrict__ gamma, const float* __restrict__ beta,
                            float* __restrict__ scale, float* __restrict__ shift) {
    int f = threadIdx.x;
    float inv_n = 1.0f / (float)NODES;
    float mean = gsum[f] * inv_n;
    float var  = gsq[f] * inv_n - mean * mean;
    float sc = gamma[f] * rsqrtf(var + BN_EPS);
    scale[f] = sc;
    shift[f] = beta[f] - mean * sc;
}

// ---------------- BN apply, in-place, float4 ----------------
__global__ void bn_apply(float* __restrict__ out, const float* __restrict__ scale,
                         const float* __restrict__ shift, long long n4) {
    long long i = (long long)blockIdx.x * blockDim.x + threadIdx.x;
    long long stride = (long long)gridDim.x * blockDim.x;
    for (; i < n4; i += stride) {
        float4 v = ((float4*)out)[i];
        int col = (int)((i * 4) & (DIM - 1));
        v.x = v.x * scale[col + 0] + shift[col + 0];
        v.y = v.y * scale[col + 1] + shift[col + 1];
        v.z = v.z * scale[col + 2] + shift[col + 2];
        v.w = v.w * scale[col + 3] + shift[col + 3];
        ((float4*)out)[i] = v;
    }
}

extern "C" void kernel_launch(void* const* d_in, const int* in_sizes, int n_in,
                              void* d_out, int out_size, void* d_ws, size_t ws_size,
                              hipStream_t stream) {
    const float* x       = (const float*)d_in[0];
    const int*   ei      = (const int*)  d_in[1];
    const float* W       = (const float*)d_in[2];
    const float* bias    = (const float*)d_in[3];
    const float* a_prelu = (const float*)d_in[4];
    const float* gamma   = (const float*)d_in[5];
    const float* beta    = (const float*)d_in[6];

    int E = in_sizes[1] / 2;
    const int* src = ei;
    const int* dst = ei + E;
    float* out = (float*)d_out;

    // workspace layout (floats)
    float* ws   = (float*)d_ws;
    float* deg  = ws;                                   // 100000
    float* dinv = ws + 100000;                          // 100000
    float* h2   = ws + 200000;                          // NODES*DIM = 12.8M
    __bf16* Wt  = (__bf16*)(ws + 200000 + (long long)NODES * DIM); // 49152 bf16
    float* gsum  = ws + 200000 + (long long)NODES * DIM + 24576;  // 128
    float* gsq   = gsum + DIM;                                    // 128
    float* scale = gsq + DIM;                                     // 128
    float* shift = scale + DIM;                                   // 128

    const long long feat4 = (long long)NODES * DIM / 4;

    // zero accumulators every call (graph replay does not re-zero)
    zero_f4<<<2048, 256, 0, stream>>>((float4*)out, feat4);
    zero_f4<<<2048, 256, 0, stream>>>((float4*)h2, feat4);
    zero_f4<<<128, 256, 0, stream>>>((float4*)deg, NODES / 4);
    zero_f4<<<1, 64, 0, stream>>>((float4*)gsum, (2 * DIM) / 4);

    deg_kernel<<<2048, 256, 0, stream>>>(src, deg, E);
    dinv_kernel<<<(NODES + 255) / 256, 256, 0, stream>>>(deg, dinv, NODES);

    // tx1 = lhat(x), staged in d_out; h2 = lhat(tx1)
    spmv_kernel<<<4096, 256, 0, stream>>>(src, dst, dinv, x, out, E);
    spmv_kernel<<<4096, 256, 0, stream>>>(src, dst, dinv, out, h2, E);

    prep_w<<<(3 * DIM * DIM + 255) / 256, 256, 0, stream>>>(W, Wt);

    // fused GEMM + bias + PReLU + stats; tx1 aliases out (safe: per-wave read-own-rows-then-write)
    int strips = NODES / 16;                 // 6250
    cheb_gemm<<<(strips + 3) / 4, 128, 0, stream>>>(x, out, h2, Wt, bias, a_prelu,
                                                    out, gsum, gsq);

    bn_finalize<<<1, DIM, 0, stream>>>(gsum, gsq, gamma, beta, scale, shift);
    bn_apply<<<2048, 256, 0, stream>>>(out, scale, shift, feat4);
}
